// MyRNNLoop_40827959116131
// MI455X (gfx1250) — matrix operand order repaired and checked by reading, not compiled, not run on hardware
//
#include <hip/hip_runtime.h>
#include <math.h>

typedef __attribute__((ext_vector_type(16))) _Float16 v16h;
typedef __attribute__((ext_vector_type(8)))  _Float16 v8h;
typedef __attribute__((ext_vector_type(8)))  float    v8f;
typedef __attribute__((ext_vector_type(4)))  int      v4i;

#define T_STEPS 256
#define BB      128
#define DD      1024
#define MM      (T_STEPS * BB)   // 32768 rows of X (flattened [T,B])
#define BKH     64               // K slab width (halves) per stage
#define PITCH   72               // padded LDS pitch: 144B rows, conflict-free

// ---- CDNA5 async global->LDS copy (ASYNCcnt) with inline-asm fallback -----
#if defined(__has_builtin)
#if __has_builtin(__builtin_amdgcn_global_load_async_to_lds_b128)
#define HAVE_ASYNC_BUILTIN 1
#endif
#if __has_builtin(__builtin_amdgcn_s_wait_asynccnt)
#define HAVE_WAIT_BUILTIN 1
#endif
#endif

#ifdef HAVE_ASYNC_BUILTIN
// Builtin parameter types are v4i* in the global(1)/local(3) address spaces.
#define ASYNC_B128(gp, lp)                                              \
  __builtin_amdgcn_global_load_async_to_lds_b128(                       \
      (__attribute__((address_space(1))) v4i*)(gp),                     \
      (__attribute__((address_space(3))) v4i*)(lp), 0, 0)
#else
#define ASYNC_B128(gp, lp)                                              \
  asm volatile("global_load_async_to_lds_b128 %0, %1, off" ::           \
                   "v"((unsigned)(unsigned long)(                       \
                       (__attribute__((address_space(3))) void*)(lp))), \
                   "v"(gp)                                              \
               : "memory")
#endif

#ifdef HAVE_WAIT_BUILTIN
#define WAIT_ASYNC(n) __builtin_amdgcn_s_wait_asynccnt(n)
#else
#define WAIT_ASYNC(n) asm volatile("s_wait_asynccnt %0" ::"n"(n) : "memory")
#endif

// ---------------------------------------------------------------------------
// Kernel 0: one-shot f32 -> f16 conversion (memory-bound prep pass) so the
// GEMM hot loop stages pure f16 via async DMA with zero VALU involvement.
// ---------------------------------------------------------------------------
__global__ __launch_bounds__(256) void cvt_f32_f16(
    const float* __restrict__ s, _Float16* __restrict__ d, int n8) {
  const int i = blockIdx.x * 256 + threadIdx.x;
  if (i >= n8) return;
  const size_t o = (size_t)i * 8;
  const float4 a = *(const float4*)(s + o);
  const float4 b = *(const float4*)(s + o + 4);
  *(v8h*)(d + o) =
      (v8h){(_Float16)a.x, (_Float16)a.y, (_Float16)a.z, (_Float16)a.w,
            (_Float16)b.x, (_Float16)b.y, (_Float16)b.z, (_Float16)b.w};
}

// ---------------------------------------------------------------------------
// Kernel 1: Z[m][n] = sum_k X[m][k]*W[n][k] + bias[n]   (f16 WMMA, f32 acc)
// Block = 256 threads (8 waves) -> 128(M) x 256(N) tile.
// Waves 2x4, each owns 64x64 = 4x4 WMMA accumulators (32 WMMA / K-slab).
// K staged in 64-wide f16 slabs, double-buffered via async global->LDS DMA.
// ---------------------------------------------------------------------------
__global__ __launch_bounds__(256) void gemm_bias_wmma(
    const _Float16* __restrict__ Xh, const _Float16* __restrict__ Wh,
    const float* __restrict__ bias, float* __restrict__ Z) {
  __shared__ _Float16 Xs[2][128][PITCH];
  __shared__ _Float16 Ws[2][256][PITCH];

  const int tid  = threadIdx.x;
  const int lane = tid & 31;
  const int wave = tid >> 5;
  const int wm   = wave >> 2;   // 0..1 : M direction (64 rows)
  const int wn   = wave & 3;    // 0..3 : N direction (64 cols)
  const int hsel = lane >> 4;   // half-wave select (K interleave)
  const int l16  = lane & 15;

  const int m0 = blockIdx.y * 128;
  const int n0 = blockIdx.x * 256;

  // staging: 8 lanes per row (16B each = 64 halves), 32 rows per pass
  const int srow = tid >> 3;        // 0..31
  const int scol = (tid & 7) * 8;   // 0,8,...,56 (halves)

  v8f acc[4][4] = {};

  auto issue_stage = [&](int buf, int k0) {
#pragma unroll
    for (int p = 0; p < 4; ++p) {   // X tile: 128 rows
      const int r = srow + p * 32;
      ASYNC_B128(Xh + (size_t)(m0 + r) * DD + k0 + scol, &Xs[buf][r][scol]);
    }
#pragma unroll
    for (int p = 0; p < 8; ++p) {   // W tile: 256 rows
      const int r = srow + p * 32;
      ASYNC_B128(Wh + (size_t)(n0 + r) * DD + k0 + scol, &Ws[buf][r][scol]);
    }
  };

  issue_stage(0, 0);                // prologue: fill buffer 0 (12 async ops)

  for (int k0 = 0; k0 < DD; k0 += BKH) {
    const int cur = (k0 >> 6) & 1;
    if (k0 + BKH < DD) {
      issue_stage(cur ^ 1, k0 + BKH);  // prefetch next slab into other buffer
      WAIT_ASYNC(12);                  // wait for current slab (12 in flight)
    } else {
      WAIT_ASYNC(0);
    }
    __syncthreads();

#pragma unroll
    for (int ks = 0; ks < BKH; ks += 32) {
      // A fragments (16-bit 16x32): elems 0..7 -> K {0..7|8..15} by half-wave,
      // elems 8..15 -> K {16..23|24..31}.
      v16h afr[4];
#pragma unroll
      for (int mi = 0; mi < 4; ++mi) {
        const _Float16* rp = &Xs[cur][wm * 64 + mi * 16 + l16][ks];
        const v8h lo = *(const v8h*)(rp + hsel * 8);
        const v8h hi = *(const v8h*)(rp + 16 + hsel * 8);
        afr[mi] = __builtin_shufflevector(lo, hi, 0, 1, 2, 3, 4, 5, 6, 7,
                                          8, 9, 10, 11, 12, 13, 14, 15);
      }
      // B fragments (32x16): lane = column N, K {0..7|8..15} then {16..23|24..31} by half-wave.
      v16h bfr[4];
#pragma unroll
      for (int ni = 0; ni < 4; ++ni) {
        const _Float16* rp = &Ws[cur][wn * 64 + ni * 16 + l16][ks];
        const v8h lo = *(const v8h*)(rp + hsel * 8);
        const v8h hi = *(const v8h*)(rp + 16 + hsel * 8);
        bfr[ni] = __builtin_shufflevector(lo, hi, 0, 1, 2, 3, 4, 5, 6, 7,
                                          8, 9, 10, 11, 12, 13, 14, 15);
      }
#pragma unroll
      for (int mi = 0; mi < 4; ++mi)
#pragma unroll
        for (int ni = 0; ni < 4; ++ni)
          acc[mi][ni] = __builtin_amdgcn_wmma_f32_16x16x32_f16(
              false, afr[mi], false, bfr[ni], (short)0, acc[mi][ni],
              false, false);
    }
    __syncthreads();   // all waves done reading before this buffer is refilled
  }

  // Epilogue: C/D f32 layout -> lane holds (M = v + 8*hsel, N = l16)
#pragma unroll
  for (int mi = 0; mi < 4; ++mi) {
#pragma unroll
    for (int ni = 0; ni < 4; ++ni) {
      const int n  = n0 + wn * 64 + ni * 16 + l16;
      const float bn = bias[n];
      const int mb = m0 + wm * 64 + mi * 16 + hsel * 8;
#pragma unroll
      for (int v = 0; v < 8; ++v)
        Z[(size_t)(mb + v) * DD + n] = acc[mi][ni][v] + bn;
    }
  }
}

// ---------------------------------------------------------------------------
// Kernel 2: per-timestep gate sign. One block per t; deterministic fixed-order
// reduction (sequential per-thread accumulation + LDS tree), no float atomics.
// ---------------------------------------------------------------------------
__global__ __launch_bounds__(256) void reduce_sign(
    const float* __restrict__ Z, float* __restrict__ sgn) {
  __shared__ float red[256];
  const int t = blockIdx.x;
  const float* zt = Z + (size_t)t * (BB * DD);
  float s = 0.f;
  for (int i = threadIdx.x; i < BB * DD; i += 256) s += zt[i];
  red[threadIdx.x] = s;
  __syncthreads();
  for (int off = 128; off > 0; off >>= 1) {
    if (threadIdx.x < off) red[threadIdx.x] += red[threadIdx.x + off];
    __syncthreads();
  }
  if (threadIdx.x == 0) sgn[t] = (red[0] > 0.f) ? 1.f : -1.f;
}

// ---------------------------------------------------------------------------
// Kernel 3: the only truly sequential part. Each thread owns one (b,d) slot:
// h = tanh(sgn[t]*Z[t] + h) over t. Coalesced 128MB stream, signs from LDS.
// ---------------------------------------------------------------------------
__global__ __launch_bounds__(256) void scan_tanh(
    const float* __restrict__ Z, const float* __restrict__ sgn,
    float* __restrict__ out) {
  __shared__ float sg[T_STEPS];
  if (threadIdx.x < T_STEPS) sg[threadIdx.x] = sgn[threadIdx.x];
  __syncthreads();
  const size_t i = (size_t)blockIdx.x * 256 + threadIdx.x;
  const float* zp = Z + i;
  float h = 0.f;
#pragma unroll 4
  for (int t = 0; t < T_STEPS; ++t) {
    h = tanhf(sg[t] * zp[0] + h);
    zp += (size_t)BB * DD;
  }
  out[i] = h;                       // reference returns (h, h)
  out[(size_t)BB * DD + i] = h;
}

// ---------------------------------------------------------------------------
extern "C" void kernel_launch(void* const* d_in, const int* in_sizes, int n_in,
                              void* d_out, int out_size, void* d_ws,
                              size_t ws_size, hipStream_t stream) {
  const float* xs = (const float*)d_in[0];   // [T,B,D] = [256,128,1024]
  const float* W  = (const float*)d_in[1];   // [D,D]
  const float* b  = (const float*)d_in[2];   // [D]
  float* out = (float*)d_out;                // 2 * B * D floats

  // workspace layout
  float*    Z   = (float*)d_ws;                        // 128 MB f32
  _Float16* Xh  = (_Float16*)(Z + (size_t)MM * DD);    // 64 MB f16
  _Float16* Wh  = Xh + (size_t)MM * DD;                // 2 MB f16
  float*    sgn = (float*)(Wh + (size_t)DD * DD);      // 1 KB

  cvt_f32_f16<<<(MM * DD / 8) / 256, 256, 0, stream>>>(xs, Xh, MM * DD / 8);
  cvt_f32_f16<<<(DD * DD / 8) / 256, 256, 0, stream>>>(W, Wh, DD * DD / 8);

  dim3 ggrid(DD / 256, MM / 128);            // (4, 256) blocks
  gemm_bias_wmma<<<ggrid, 256, 0, stream>>>(Xh, Wh, b, Z);
  reduce_sign<<<T_STEPS, 256, 0, stream>>>(Z, sgn);
  scan_tanh<<<(BB * DD) / 256, 256, 0, stream>>>(Z, sgn, out);
}
